// AttentionReadout_9929964388802
// MI455X (gfx1250) — compile-verified, matrix operand order checked by reading, not run on hardware
//
#include <hip/hip_runtime.h>
#include <hip/hip_bf16.h>

typedef __attribute__((ext_vector_type(16))) __bf16 v16bf;
typedef __attribute__((ext_vector_type(8)))  float  v8f;

#define FDIM 128
#define HIDD 32
#define NH   4

__device__ __forceinline__ float siluf(float x) {
    return x / (1.0f + __expf(-x));
}
// order-preserving float <-> uint map (for deterministic atomicMax on floats)
__device__ __forceinline__ unsigned ford(float f) {
    unsigned u = __float_as_uint(f);
    return (u & 0x80000000u) ? ~u : (u | 0x80000000u);
}
__device__ __forceinline__ float funord(unsigned u) {
    return __uint_as_float((u & 0x80000000u) ? (u & 0x7fffffffu) : ~u);
}

// ---------------- Phase 1: logits = silu(X@W1+b1)@W2+b2 ------------------
// One wave handles a 16-atom tile. W1 kept resident as bf16 B-fragments.
__global__ __launch_bounds__(128) void k_logits(
    const float* __restrict__ X, const float* __restrict__ W1,
    const float* __restrict__ b1, const float* __restrict__ W2,
    const float* __restrict__ b2, float* __restrict__ lg, int N)
{
    __shared__ float hid[4][16][HIDD + 1];   // per-wave slice
    const int lane = threadIdx.x & 31;
    const int wave = threadIdx.x >> 5;
    const int r    = lane & 15;
    const int half = lane >> 4;              // 0: K {0-7,16-23}  1: K {8-15,24-31}
    const int koff = half * 8;

    // Pre-pack W1 into B-fragments: Bf[kchunk][ntile] (held in VGPRs)
    v16bf Bf[4][2];
#pragma unroll
    for (int kc = 0; kc < 4; ++kc)
#pragma unroll
        for (int nt = 0; nt < 2; ++nt) {
            const int n  = r + nt * 16;
            const int kb = kc * 32 + koff;
#pragma unroll
            for (int j = 0; j < 8; ++j) {
                Bf[kc][nt][j]     = (__bf16)W1[(kb + j)      * HIDD + n];
                Bf[kc][nt][8 + j] = (__bf16)W1[(kb + 16 + j) * HIDD + n];
            }
        }
    const float b1n0 = b1[r];
    const float b1n1 = b1[r + 16];
    // W2 columns for this lane's two heads (h = half, half+2), resident
    float w2a[HIDD], w2b[HIDD];
#pragma unroll
    for (int k = 0; k < HIDD; ++k) {
        w2a[k] = W2[k * NH + half];
        w2b[k] = W2[k * NH + half + 2];
    }
    const float b2a = b2[half], b2b = b2[half + 2];

    const int tiles = (N + 15) >> 4;
    const int nwaves = gridDim.x * 4;
    for (int tile = blockIdx.x * 4 + wave; tile < tiles; tile += nwaves) {
        const int abase = tile * 16;
        int arow = abase + r; if (arow >= N) arow = N - 1;
        const float* xrow = X + (size_t)arow * FDIM;

        v8f c0 = {}; v8f c1 = {};
#pragma unroll
        for (int kc = 0; kc < 4; ++kc) {
            const int kb = kc * 32 + koff;
            v16bf a;
#pragma unroll
            for (int j = 0; j < 8; ++j) {
                a[j]     = (__bf16)xrow[kb + j];
                a[8 + j] = (__bf16)xrow[kb + 16 + j];
            }
            c0 = __builtin_amdgcn_wmma_f32_16x16x32_bf16(false, a, false, Bf[kc][0],
                                                         (short)0, c0, false, false);
            c1 = __builtin_amdgcn_wmma_f32_16x16x32_bf16(false, a, false, Bf[kc][1],
                                                         (short)0, c1, false, false);
        }
        // silu(hidden) -> per-wave LDS tile (C layout: elem j -> M=j+8*half, N=r)
#pragma unroll
        for (int j = 0; j < 8; ++j) {
            const int m = j + half * 8;
            hid[wave][m][r]      = siluf(c0[j] + b1n0);
            hid[wave][m][r + 16] = siluf(c1[j] + b1n1);
        }
        // intra-wave LDS RAW: wait for DS writes (wave is lockstep, no barrier needed)
        asm volatile("s_wait_dscnt 0x0" ::: "memory");

        // lane computes logits for (row r, heads half and half+2)
        float acc0 = b2a, acc1 = b2b;
#pragma unroll
        for (int k = 0; k < HIDD; ++k) {
            const float hv = hid[wave][r][k];
            acc0 = fmaf(hv, w2a[k], acc0);
            acc1 = fmaf(hv, w2b[k], acc1);
        }
        const int arow2 = abase + r;
        if (arow2 < N) {
            lg[(size_t)arow2 * NH + half]     = acc0;
            lg[(size_t)arow2 * NH + half + 2] = acc1;
        }
    }
}

// ---------------- Phase 2: segment max (deterministic atomics) -----------
__global__ void k_init(unsigned* __restrict__ smax, int BH) {
    const int t = blockIdx.x * blockDim.x + threadIdx.x;
    if (t < BH) smax[t] = 0x007FFFFFu;   // ford(-inf)
}

__global__ void k_segmax(const float* __restrict__ lg, const int* __restrict__ seg,
                         unsigned* __restrict__ smax, int N)
{
    const int t = blockIdx.x * blockDim.x + threadIdx.x;
    const int base = t * 8;
    if (base >= N) return;
    const int end = min(base + 8, N);
    int curb = seg[base];
    float m[NH] = {-3.4e38f, -3.4e38f, -3.4e38f, -3.4e38f};
    for (int a = base; a < end; ++a) {
        const int b = seg[a];
        if (b != curb) {
#pragma unroll
            for (int h = 0; h < NH; ++h) {
                atomicMax(&smax[curb * NH + h], ford(m[h]));
                m[h] = -3.4e38f;
            }
            curb = b;
        }
#pragma unroll
        for (int h = 0; h < NH; ++h)
            m[h] = fmaxf(m[h], lg[(size_t)a * NH + h]);
    }
#pragma unroll
    for (int h = 0; h < NH; ++h)
        atomicMax(&smax[curb * NH + h], ford(m[h]));
}

// ---------------- Phase 3: ex = exp(logit - segmax) ----------------------
__global__ void k_exp(const float* __restrict__ lg, const int* __restrict__ seg,
                      const unsigned* __restrict__ smax, float* __restrict__ ex, int N)
{
    const int a = blockIdx.x * blockDim.x + threadIdx.x;
    if (a >= N) return;
    const int b = seg[a];
#pragma unroll
    for (int h = 0; h < NH; ++h) {
        const float mx = funord(smax[b * NH + h]);
        ex[(size_t)a * NH + h] = __expf(lg[(size_t)a * NH + h] - mx);
    }
}

// ---------------- Phase 4: per-crystal softmax-weighted sum --------------
// One block per crystal; bounds via binary search on sorted segment ids.
// Denominator via fixed-tree block reduction (deterministic, no fp atomics).
__global__ __launch_bounds__(256) void k_weighted(
    const float* __restrict__ X, const int* __restrict__ seg,
    const float* __restrict__ ex, float* __restrict__ wsum, int N)
{
    __shared__ float xs[8][FDIM];
    __shared__ float wl[8][NH];
    __shared__ float red[NH][256];
    __shared__ float dinv[NH];

    const int b = blockIdx.x;
    const int t = threadIdx.x;
    // [s, e) = range of atoms with seg==b
    int lo = 0, hi = N;
    while (lo < hi) { int mid = (lo + hi) >> 1; if (seg[mid] <  b) lo = mid + 1; else hi = mid; }
    const int s = lo;
    hi = N;
    while (lo < hi) { int mid = (lo + hi) >> 1; if (seg[mid] <= b) lo = mid + 1; else hi = mid; }
    const int e = lo;

    // denominator: deterministic tree reduction of ex over the segment
    float p[NH] = {0.f, 0.f, 0.f, 0.f};
    for (int a = s + t; a < e; a += 256) {
#pragma unroll
        for (int h = 0; h < NH; ++h) p[h] += ex[(size_t)a * NH + h];
    }
#pragma unroll
    for (int h = 0; h < NH; ++h) red[h][t] = p[h];
    __syncthreads();
    for (int off = 128; off > 0; off >>= 1) {
        if (t < off) {
#pragma unroll
            for (int h = 0; h < NH; ++h) red[h][t] += red[h][t + off];
        }
        __syncthreads();
    }
    if (t < NH) {
        const float d = red[t][0];
        dinv[t] = (d != 0.f) ? 1.f / d : 0.f;
    }

    const int f  = t & 127;
    const int h1 = t >> 7;               // 0 or 1; thread owns heads h1 and h1+2
    float acc0 = 0.f, acc1 = 0.f;

    for (int c = s; c < e; c += 8) {
        const int cnt = min(8, e - c);
        __syncthreads();                  // protect previous chunk + dinv visibility
        for (int i = t; i < cnt * FDIM; i += 256)
            xs[i >> 7][i & 127] = X[(size_t)(c + (i >> 7)) * FDIM + (i & 127)];
        if (t < cnt * NH) {
            const int a = t >> 2, h = t & 3;
            wl[a][h] = ex[(size_t)(c + a) * NH + h] * dinv[h];
        }
        __syncthreads();
        for (int a = 0; a < cnt; ++a) {
            const float xv = xs[a][f];
            acc0 = fmaf(wl[a][h1],     xv, acc0);
            acc1 = fmaf(wl[a][h1 + 2], xv, acc1);
        }
    }
    wsum[(size_t)b * (NH * FDIM) + h1 * FDIM + f]       = acc0;
    wsum[(size_t)b * (NH * FDIM) + (h1 + 2) * FDIM + f] = acc1;
}

// ---------------- Phase 5: out = silu((B,512)@Wp + bp), bf16 WMMA --------
__global__ __launch_bounds__(256) void k_proj(
    const float* __restrict__ wsum, const float* __restrict__ Wp,
    const float* __restrict__ bp, float* __restrict__ out, int B)
{
    const int lane = threadIdx.x & 31;
    const int wave = threadIdx.x >> 5;    // n-tile 0..7
    const int r    = lane & 15;
    const int half = lane >> 4;
    const int koff = half * 8;
    const int rbase = blockIdx.x * 16;
    const int n = wave * 16 + r;          // output feature

    int row = rbase + r; if (row >= B) row = B - 1;
    const float* arow = wsum + (size_t)row * (NH * FDIM);

    v8f c = {};
    for (int kc = 0; kc < NH * FDIM; kc += 32) {
        const int kb = kc + koff;
        v16bf a, bfr;
#pragma unroll
        for (int j = 0; j < 8; ++j) {
            a[j]       = (__bf16)arow[kb + j];
            a[8 + j]   = (__bf16)arow[kb + 16 + j];
            bfr[j]     = (__bf16)Wp[(size_t)(kb + j)      * FDIM + n];
            bfr[8 + j] = (__bf16)Wp[(size_t)(kb + 16 + j) * FDIM + n];
        }
        c = __builtin_amdgcn_wmma_f32_16x16x32_bf16(false, a, false, bfr,
                                                    (short)0, c, false, false);
    }
    const float bpn = bp[n];
#pragma unroll
    for (int j = 0; j < 8; ++j) {
        const int m = rbase + j + half * 8;
        if (m < B) out[(size_t)m * FDIM + n] = siluf(c[j] + bpn);
    }
}

// -------------------------------------------------------------------------
extern "C" void kernel_launch(void* const* d_in, const int* in_sizes, int n_in,
                              void* d_out, int out_size, void* d_ws, size_t ws_size,
                              hipStream_t stream)
{
    if (n_in < 9) return;
    const float* X   = (const float*)d_in[0];
    const int*   seg = (const int*)  d_in[1];
    const float* W1  = (const float*)d_in[3];
    const float* b1  = (const float*)d_in[4];
    const float* W2  = (const float*)d_in[5];
    const float* b2  = (const float*)d_in[6];
    const float* Wp  = (const float*)d_in[7];
    const float* bp  = (const float*)d_in[8];

    const int N  = in_sizes[1];          // atoms
    const int Bc = out_size / FDIM;      // crystals

    // workspace partition (floats / uints, 4B each)
    float*    ws    = (float*)d_ws;
    float*    lg    = ws;                          // N*4
    float*    ex    = lg + (size_t)N * NH;         // N*4
    unsigned* smax  = (unsigned*)(ex + (size_t)N * NH);  // Bc*4
    float*    wsum  = (float*)(smax + (size_t)Bc * NH);  // Bc*512

    const int tiles = (N + 15) >> 4;
    int blocks1 = (tiles + 3) >> 2;
    if (blocks1 > 1024) blocks1 = 1024;
    k_logits<<<blocks1, 128, 0, stream>>>(X, W1, b1, W2, b2, lg, N);

    const int BH = Bc * NH;
    k_init<<<(BH + 255) / 256, 256, 0, stream>>>(smax, BH);

    const int segThreads = (N + 7) / 8;
    k_segmax<<<(segThreads + 255) / 256, 256, 0, stream>>>(lg, seg, smax, N);

    k_exp<<<(N + 255) / 256, 256, 0, stream>>>(lg, seg, smax, ex, N);

    k_weighted<<<Bc, 256, 0, stream>>>(X, seg, ex, wsum, N);

    k_proj<<<(Bc + 15) / 16, 256, 0, stream>>>(wsum, Wp, bp, (float*)d_out, Bc);
}
